// GNNLayer_67912022884494
// MI455X (gfx1250) — matrix operand order repaired and checked by reading, not compiled, not run on hardware
//
#include <hip/hip_runtime.h>
#include <hip/hip_bf16.h>

typedef _Float16 half_t;
typedef __attribute__((ext_vector_type(16))) _Float16 v16h;
typedef __attribute__((ext_vector_type(8)))  _Float16 v8h;
typedef __attribute__((ext_vector_type(8)))  float    v8f;

#define C_DIM 256
#define HEADS 4
#define DHEAD 64
#define NEG_SLOPE 0.2f
#define BN_EPS 1e-5f

// ---------------------------------------------------------------- utilities
__global__ void k_fill(float* __restrict__ p, float v, int n) {
    int i = blockIdx.x * blockDim.x + threadIdx.x;
    if (i < n) p[i] = v;
}

__global__ void k_cvt_f16(const float* __restrict__ src, half_t* __restrict__ dst, int n) {
    int i = blockIdx.x * blockDim.x + threadIdx.x;
    if (i < n) dst[i] = (half_t)src[i];
}

// ---------------------------------------------------------------- degrees
__global__ void k_deg(const int* __restrict__ ei, const float* __restrict__ ew,
                      float* __restrict__ deg, int E) {
    int e = blockIdx.x * blockDim.x + threadIdx.x;
    if (e < E) atomicAdd(&deg[ei[E + e]], ew[e]);
}

__global__ void k_dinv(const float* __restrict__ deg, float* __restrict__ dinv, int n) {
    int i = blockIdx.x * blockDim.x + threadIdx.x;
    if (i < n) {
        float d = deg[i];
        dinv[i] = d > 0.f ? rsqrtf(fmaxf(d, 1e-12f)) : 0.f;
    }
}

__global__ void k_norm(const int* __restrict__ ei, const float* __restrict__ ew,
                       const float* __restrict__ dinv, float* __restrict__ nrm,
                       int E, int N) {
    int e = blockIdx.x * blockDim.x + threadIdx.x;
    if (e >= E + N) return;
    int s, d; float w;
    if (e < E) { s = ei[e]; d = ei[E + e]; w = ew[e]; }
    else       { s = d = e - E; w = 1.f; }
    nrm[e] = dinv[s] * w * dinv[d];
}

// ---------------------------------------------------------------- WMMA GEMM
// C[nrows x 256] = A[nrows x 256](f16, row-major) * B[256 x 256](f16, row-major)
// block = 512 threads = 16 waves; wave w -> column tile n0 = 16*w; block -> 16 rows.
__global__ void __launch_bounds__(512)
k_gemm_wmma(const half_t* __restrict__ A, const half_t* __restrict__ B,
            float* __restrict__ Cout, int nrows) {
    const int wave = threadIdx.x >> 5;
    const int lane = threadIdx.x & 31;
    const int m0   = blockIdx.x * 16;
    const int n0   = wave * 16;
    const int hi   = (lane >> 4) & 1;              // lanes 16-31 -> upper K/M halves
    int row = m0 + (lane & 15);
    if (row >= nrows) row = 0;                     // clamp for tail blocks (loads only)

    const half_t* arow = A + (size_t)row * C_DIM + hi * 8;
    v8f acc = {};
#pragma unroll
    for (int kb = 0; kb < C_DIM; kb += 32) {
        // A fragment: two contiguous 8-half chunks per ISA 16-bit A layout
        v8h a_lo = *(const v8h*)(arow + kb);
        v8h a_hi = *(const v8h*)(arow + kb + 16);
        v16h a = __builtin_shufflevector(a_lo, a_hi,
                                         0, 1, 2, 3, 4, 5, 6, 7,
                                         8, 9, 10, 11, 12, 13, 14, 15);
        // B fragment: lane holds column n0 + (lane&15), K in same order as A
        v16h b;
#pragma unroll
        for (int s = 0; s < 8; ++s) {
            b[s]     = B[(size_t)(kb + hi * 8 + s) * C_DIM + n0 + (lane & 15)];
            b[s + 8] = B[(size_t)(kb + 16 + hi * 8 + s) * C_DIM + n0 + (lane & 15)];
        }
        if (kb + 32 < C_DIM)                        // gfx1250 global_prefetch_b8
            __builtin_prefetch((const void*)(B + (size_t)(kb + 32 + lane) * C_DIM + n0), 0, 1);
        acc = __builtin_amdgcn_wmma_f32_16x16x32_f16(
            false, a, false, b, (short)0, acc, false, false);
    }
    // C/D layout: VGPR r -> row m0 + r + 8*hi, col n0 + (lane&15)
    const int col = n0 + (lane & 15);
#pragma unroll
    for (int r = 0; r < 8; ++r) {
        int rg = m0 + r + hi * 8;
        if (rg < nrows) Cout[(size_t)rg * C_DIM + col] = acc[r];
    }
}

// ---------------------------------------------------------------- GAT alphas
__global__ void k_alpha(const float* __restrict__ hg, const float* __restrict__ a_src,
                        const float* __restrict__ a_dst, float* __restrict__ als,
                        float* __restrict__ ald, int N) {
    int i = blockIdx.x * blockDim.x + threadIdx.x;
    if (i >= N * HEADS) return;
    int n = i >> 2, hh = i & 3;
    const float* rowp = hg + (size_t)n * C_DIM + hh * DHEAD;
    const float* as = a_src + hh * DHEAD;
    const float* ad = a_dst + hh * DHEAD;
    float s = 0.f, d = 0.f;
#pragma unroll 8
    for (int k = 0; k < DHEAD; ++k) {
        float v = rowp[k];
        s += v * as[k];
        d += v * ad[k];
    }
    als[i] = s; ald[i] = d;
}

// ---------------------------------------------------------------- GCN scatter
__global__ void k_gcn_agg(const int* __restrict__ ei, const float* __restrict__ nrm,
                          const float* __restrict__ h, float* __restrict__ outp,
                          int E, int N) {
    int idx = blockIdx.x * blockDim.x + threadIdx.x;   // (E+N)*64 items (float4 each)
    int total = (E + N) * (C_DIM / 4);
    if (idx >= total) return;
    int e = idx >> 6;
    int c = (idx & 63) << 2;
    int s, d;
    if (e < E) { s = ei[e]; d = ei[E + e]; }
    else       { s = d = e - E; }
    float w = nrm[e];
    float4 v = *(const float4*)(h + (size_t)s * C_DIM + c);
    float* o = outp + (size_t)d * C_DIM + c;
    atomicAdd(o + 0, w * v.x);
    atomicAdd(o + 1, w * v.y);
    atomicAdd(o + 2, w * v.z);
    atomicAdd(o + 3, w * v.w);
}

// ---------------------------------------------------------------- attention weights
__global__ void k_ee(const int* __restrict__ ei, const float* __restrict__ als,
                     const float* __restrict__ ald, float* __restrict__ ee,
                     float* __restrict__ esum, int E, int N) {
    int i = blockIdx.x * blockDim.x + threadIdx.x;     // (E+N)*HEADS items
    if (i >= (E + N) * HEADS) return;
    int e = i >> 2, hh = i & 3;
    int s, d;
    if (e < E) { s = ei[e]; d = ei[E + e]; }
    else       { s = d = e - E; }
    float v = als[s * HEADS + hh] + ald[d * HEADS + hh];
    v = v > 0.f ? v : NEG_SLOPE * v;                   // leaky_relu
    float w = __expf(v);                               // softmax is shift-invariant
    ee[i] = w;
    atomicAdd(&esum[d * HEADS + hh], w);
}

// ---------------------------------------------------------------- GAT scatter
__global__ void k_gat_agg(const int* __restrict__ ei, const float* __restrict__ ee,
                          const float* __restrict__ hg, float* __restrict__ outp,
                          int E, int N) {
    int idx = blockIdx.x * blockDim.x + threadIdx.x;
    int total = (E + N) * (C_DIM / 4);
    if (idx >= total) return;
    int e = idx >> 6;
    int c = (idx & 63) << 2;
    int s, d;
    if (e < E) { s = ei[e]; d = ei[E + e]; }
    else       { s = d = e - E; }
    float w = ee[e * HEADS + (c >> 6)];
    float4 v = *(const float4*)(hg + (size_t)s * C_DIM + c);
    float* o = outp + (size_t)d * C_DIM + c;
    atomicAdd(o + 0, w * v.x);
    atomicAdd(o + 1, w * v.y);
    atomicAdd(o + 2, w * v.z);
    atomicAdd(o + 3, w * v.w);
}

// ---------------------------------------------------------------- normalize + bias -> f16
__global__ void k_gat_fin(const float* __restrict__ gat_agg, const float* __restrict__ esum,
                          const float* __restrict__ b_gat, half_t* __restrict__ gath, int N) {
    int i = blockIdx.x * blockDim.x + threadIdx.x;
    if (i >= N * C_DIM) return;
    int n = i >> 8, c = i & 255;
    float s = esum[n * HEADS + (c >> 6)];
    gath[i] = (half_t)(gat_agg[i] / s + b_gat[c]);
}

// ---------------------------------------------------------------- epilogue
__global__ void k_final(const float* __restrict__ gcn, const float* __restrict__ proj,
                        const float* __restrict__ x, const float* __restrict__ b_gcn,
                        const float* __restrict__ b_proj, const float* __restrict__ gamma,
                        const float* __restrict__ beta, const float* __restrict__ mean,
                        const float* __restrict__ var, float* __restrict__ out, int N) {
    int i = blockIdx.x * blockDim.x + threadIdx.x;
    if (i >= N * C_DIM) return;
    int c = i & 255;
    float t = gcn[i] + b_gcn[c] + proj[i] + b_proj[c];
    t = gamma[c] * (t - mean[c]) * rsqrtf(var[c] + BN_EPS) + beta[c];
    t = t > 0.f ? t : 0.f;                              // ReLU
    out[i] = t + x[i];                                  // residual
}

// ================================================================ host
extern "C" void kernel_launch(void* const* d_in, const int* in_sizes, int n_in,
                              void* d_out, int out_size, void* d_ws, size_t ws_size,
                              hipStream_t stream) {
    (void)n_in; (void)out_size; (void)ws_size;
    const float* x      = (const float*)d_in[0];
    const int*   ei     = (const int*)d_in[1];
    const float* ew     = (const float*)d_in[2];
    const float* W_gcn  = (const float*)d_in[3];
    const float* b_gcn  = (const float*)d_in[4];
    const float* W_gat  = (const float*)d_in[5];
    const float* a_src  = (const float*)d_in[6];
    const float* a_dst  = (const float*)d_in[7];
    const float* b_gat  = (const float*)d_in[8];
    const float* W_proj = (const float*)d_in[9];
    const float* b_proj = (const float*)d_in[10];
    const float* gamma  = (const float*)d_in[11];
    const float* beta   = (const float*)d_in[12];
    const float* mean   = (const float*)d_in[13];
    const float* var    = (const float*)d_in[14];

    const int N  = in_sizes[0] / C_DIM;      // 10000
    const int E  = in_sizes[1] / 2;          // 320000
    const int ET = E + N;                    // edges + self-loops
    const int NC = N * C_DIM;

    // workspace carve-up (256B aligned)
    char* ws = (char*)d_ws;
    size_t off = 0;
    auto carve = [&](size_t bytes) -> char* {
        char* p = ws + off;
        off = (off + bytes + 255) & ~(size_t)255;
        return p;
    };
    half_t* xh      = (half_t*)carve((size_t)NC * 2);
    half_t* wgcn_h  = (half_t*)carve((size_t)C_DIM * C_DIM * 2);
    half_t* wgat_h  = (half_t*)carve((size_t)C_DIM * C_DIM * 2);
    half_t* wproj_h = (half_t*)carve((size_t)C_DIM * C_DIM * 2);
    float*  h       = (float*)carve((size_t)NC * 4);   // also reused as proj
    float*  hg      = (float*)carve((size_t)NC * 4);
    float*  gcn_agg = (float*)carve((size_t)NC * 4);
    float*  gat_agg = (float*)carve((size_t)NC * 4);
    half_t* gath    = (half_t*)carve((size_t)NC * 2);
    float*  deg     = (float*)carve((size_t)N * 4);
    float*  dinv    = (float*)carve((size_t)N * 4);
    float*  als     = (float*)carve((size_t)N * HEADS * 4);
    float*  ald     = (float*)carve((size_t)N * HEADS * 4);
    float*  esum    = (float*)carve((size_t)N * HEADS * 4);
    float*  ee      = (float*)carve((size_t)ET * HEADS * 4);
    float*  nrm     = (float*)carve((size_t)ET * 4);
    float*  proj    = h;                               // alias: h dead after k_gcn_agg

    const int T = 256;
    auto blocks = [&](int n) { return (n + T - 1) / T; };

    // prep: f16 copies + accumulator init
    k_cvt_f16<<<blocks(NC), T, 0, stream>>>(x, xh, NC);
    k_cvt_f16<<<blocks(C_DIM * C_DIM), T, 0, stream>>>(W_gcn,  wgcn_h,  C_DIM * C_DIM);
    k_cvt_f16<<<blocks(C_DIM * C_DIM), T, 0, stream>>>(W_gat,  wgat_h,  C_DIM * C_DIM);
    k_cvt_f16<<<blocks(C_DIM * C_DIM), T, 0, stream>>>(W_proj, wproj_h, C_DIM * C_DIM);
    k_fill<<<blocks(N), T, 0, stream>>>(deg, 1.0f, N);             // self-loop weight
    k_fill<<<blocks(NC), T, 0, stream>>>(gcn_agg, 0.0f, NC);
    k_fill<<<blocks(NC), T, 0, stream>>>(gat_agg, 0.0f, NC);
    k_fill<<<blocks(N * HEADS), T, 0, stream>>>(esum, 0.0f, N * HEADS);

    // degrees / symmetric norm
    k_deg<<<blocks(E), T, 0, stream>>>(ei, ew, deg, E);
    k_dinv<<<blocks(N), T, 0, stream>>>(deg, dinv, N);
    k_norm<<<blocks(ET), T, 0, stream>>>(ei, ew, dinv, nrm, E, N);

    // GEMMs on WMMA: h = x@W_gcn, hg = x@W_gat
    const int gemm_grid = (N + 15) / 16;
    k_gemm_wmma<<<gemm_grid, 512, 0, stream>>>(xh, wgcn_h, h, N);
    k_gemm_wmma<<<gemm_grid, 512, 0, stream>>>(xh, wgat_h, hg, N);

    // attention alphas
    k_alpha<<<blocks(N * HEADS), T, 0, stream>>>(hg, a_src, a_dst, als, ald, N);

    // GCN aggregation (consumes h)
    k_gcn_agg<<<blocks(ET * (C_DIM / 4)), T, 0, stream>>>(ei, nrm, h, gcn_agg, E, N);

    // GAT: edge weights + aggregation + normalize
    k_ee<<<blocks(ET * HEADS), T, 0, stream>>>(ei, als, ald, ee, esum, E, N);
    k_gat_agg<<<blocks(ET * (C_DIM / 4)), T, 0, stream>>>(ei, ee, hg, gat_agg, E, N);
    k_gat_fin<<<blocks(NC), T, 0, stream>>>(gat_agg, esum, b_gat, gath, N);

    // proj = gat@W_proj (reuses h buffer)
    k_gemm_wmma<<<gemm_grid, 512, 0, stream>>>(gath, wproj_h, proj, N);

    // BN + ReLU + residual
    k_final<<<blocks(NC), T, 0, stream>>>(gcn_agg, proj, x, b_gcn, b_proj,
                                          gamma, beta, mean, var, (float*)d_out, N);
}
